// MultiHeadAttention_85229331022244
// MI455X (gfx1250) — matrix operand order repaired and checked by reading, not compiled, not run on hardware
//
#include <hip/hip_runtime.h>

#define D_MODEL 1024
#define N_HEADS 16
#define HEAD_D  64
#define BATCH   4
#define SEQ     4096

typedef _Float16 h16;
typedef __attribute__((ext_vector_type(16))) _Float16 v16h;
typedef __attribute__((ext_vector_type(8)))  _Float16 v8h;
typedef __attribute__((ext_vector_type(8)))  float    v8f;
typedef __attribute__((ext_vector_type(4)))  unsigned int v4u;
typedef __attribute__((ext_vector_type(8)))  int      v8i;
typedef __attribute__((ext_vector_type(4)))  int      v4i;

__device__ __forceinline__ v8f wmma_f16(v16h a, v16h b, v8f c) {
  // D = A(16x32 f16) * B(32x16 f16) + C(16x16 f32)
  return __builtin_amdgcn_wmma_f32_16x16x32_f16(
      /*neg_a=*/false, a, /*neg_b=*/false, b,
      /*c_mod=*/(short)0, c, /*reuse_a=*/false, /*reuse_b=*/false);
}

// ---------------------------------------------------------------------------
// Tensor Data Mover: 2-D tile (rows x elems, row pitch = stride elements)
// global -> LDS.  Descriptor per CDNA5 ISA ch.8; groups 2/3 zero => 2-D.
// dsz_code: 0=1B, 1=2B, 2=4B elements.
// ---------------------------------------------------------------------------
__device__ __forceinline__ void tdm_load_to_lds(unsigned lds_off, const void* gptr,
                                                unsigned rows, unsigned elems,
                                                unsigned stride, unsigned dsz_code) {
  unsigned long long ga = (unsigned long long)(uintptr_t)gptr;
  v4u g0;
  g0[0] = 1u;                                         // count=1, user mode
  g0[1] = lds_off;                                    // lds_addr [63:32]
  g0[2] = (unsigned)(ga & 0xFFFFFFFFu);               // global_addr[31:0]
  g0[3] = (unsigned)((ga >> 32) & 0x01FFFFFFu)        // global_addr[56:32]
        | (2u << 30);                                 // type=2 ("image")
  const unsigned dim0  = elems;                       // tensor_dim0 (elements)
  const unsigned dim1  = rows;                        // tensor_dim1
  const unsigned tile0 = elems;                       // tile_dim0
  const unsigned tile1 = rows;                        // tile_dim1
  const unsigned long long str0 = stride;             // tensor_dim0_stride
  v8i g1;
  g1[0] = (int)(dsz_code << 16);                      // data_size
  g1[1] = (int)((dim0 & 0xFFFFu) << 16);              // dim0[15:0] @ bits 63:48
  g1[2] = (int)((dim0 >> 16) | ((dim1 & 0xFFFFu) << 16));
  g1[3] = (int)((dim1 >> 16) | (tile0 << 16));        // tile_dim0 @ 127:112
  g1[4] = (int)(tile1 & 0xFFFFu);                     // tile_dim1; tile_dim2=0
  g1[5] = (int)(str0 & 0xFFFFFFFFull);                // stride0[31:0]
  g1[6] = (int)((str0 >> 32) & 0xFFFFull);            // stride0[47:32]
  g1[7] = 0;
  v4i z4 = {0, 0, 0, 0};
  v8i z8 = {0, 0, 0, 0, 0, 0, 0, 0};
  __builtin_amdgcn_tensor_load_to_lds(g0, g1, z4, z4, z8, /*cpol=*/0);
}

// ---------------------------------------------------------------------------
// Kernel 0: repack W_{q,k,v} (f32, [qq][d][h] strided) into f16 tiles laid out
// exactly as the WMMA B operand, so the projection kernel's B load is one
// contiguous 32-byte v16h per lane.  Linear index:
//   ((((mat*16 + h)*2 + kc)*4 + nc)*32 + lane)*16 + j
// maps to  w[((kc*32 + (lane>>4)*16 + j)*64 + nc*16 + (lane&15))*16 + h].
// ---------------------------------------------------------------------------
__global__ __launch_bounds__(256) void pack_w_kernel(
    const float* __restrict__ wq, const float* __restrict__ wk,
    const float* __restrict__ wv, h16* __restrict__ wp) {
  const unsigned idx  = blockIdx.x * 256u + threadIdx.x;   // < 3*65536
  const unsigned j    = idx & 15u;
  const unsigned lane = (idx >> 4) & 31u;
  const unsigned nc   = (idx >> 9) & 3u;
  const unsigned kc   = (idx >> 11) & 1u;
  const unsigned h    = (idx >> 12) & 15u;
  const unsigned mat  = idx >> 16;
  const unsigned qq   = kc * 32u + (lane >> 4) * 16u + j;
  const unsigned d    = nc * 16u + (lane & 15u);
  const float* w = (mat == 0) ? wq : ((mat == 1) ? wk : wv);
  wp[idx] = (h16)w[((size_t)qq * HEAD_D + d) * N_HEADS + h];
}

// ---------------------------------------------------------------------------
// Kernel 1: per-head QKV projection (WMMA f16) + fused RoPE, emit f16 Q/K
// (row-major [t][d]) and V transposed ([d][t], so the attention PV B-operand
// is lane-contiguous).  Workgroup = 16 tokens, 8 waves, wave w handles heads
// {2w, 2w+1} so the RoPE channel pair (d*16+2w, d*16+2w+1) is wave-resident.
// X tile staged by TDM into LDS; outputs staged through LDS for coalescing.
// ---------------------------------------------------------------------------
__global__ __launch_bounds__(256) void qkv_rope_kernel(
    const float* __restrict__ x, const h16* __restrict__ wp,
    h16* __restrict__ qo, h16* __restrict__ ko, h16* __restrict__ vT) {
  __shared__ float ldsX[16 * D_MODEL];   // 64 KB x-tile; reused as out staging

  const int blk   = blockIdx.x;
  const int b     = blk / (SEQ / 16);
  const int t0    = (blk % (SEQ / 16)) * 16;
  const int lane  = threadIdx.x & 31;
  const int wave  = threadIdx.x >> 5;
  const int group = lane >> 4;
  const int mrow  = lane & 15;
  const int h0    = wave * 2;

  if (wave == 0) {
    tdm_load_to_lds((unsigned)(uintptr_t)&ldsX[0],
                    x + ((size_t)(b * SEQ + t0)) * D_MODEL,
                    /*rows=*/16, /*elems=*/D_MODEL, /*stride=*/D_MODEL, /*4B*/ 2);
    __builtin_amdgcn_s_wait_tensorcnt(0);
  }
  __syncthreads();

  // A tiles from LDS: X_h (16 tokens x 64 qq-channels), element x[t, qq*16+h].
  // A 16x32 tiles in this code's contiguous k order (same as packed B): slot j -> K = g*16+j.
  v16h ax[2][2];
  for (int hh = 0; hh < 2; ++hh)
    for (int kc = 0; kc < 2; ++kc) {
      v16h a;
#pragma unroll
      for (int j = 0; j < 16; ++j) {
        const int K  = group * 16 + j;
        const int qq = kc * 32 + K;
        a[j] = (h16)ldsX[mrow * D_MODEL + qq * 16 + (h0 + hh)];
      }
      ax[hh][kc] = a;
    }

  h16* ldsO = (h16*)ldsX;

  for (int mat = 0; mat < 3; ++mat) {
    v8f acc[2][4];
    for (int hh = 0; hh < 2; ++hh)
      for (int nc = 0; nc < 4; ++nc)
        acc[hh][nc] = (v8f){0.f, 0.f, 0.f, 0.f, 0.f, 0.f, 0.f, 0.f};

    for (int hh = 0; hh < 2; ++hh)
      for (int kc = 0; kc < 2; ++kc)
        for (int nc = 0; nc < 4; ++nc) {
          const v16h bt = *(const v16h*)(
              wp + ((((size_t)mat * 16 + (h0 + hh)) * 2 + kc) * 4 + nc) * 512
                 + lane * 16);
          acc[hh][nc] = wmma_f16(ax[hh][kc], bt, acc[hh][nc]);
        }

    if (mat < 2) {  // RoPE on Q and K: pair index i = d*8 + wave
      for (int nc = 0; nc < 4; ++nc) {
        const int   d    = nc * 16 + mrow;
        const float freq = __powf(10000.0f, -(float)(d * 8 + wave) * (1.0f / 512.0f));
#pragma unroll
        for (int r = 0; r < 8; ++r) {
          const int tok = t0 + r + 8 * group;   // C layout: M = r + 8*(lane/16)
          float sn, cs;
          __sincosf((float)tok * freq, &sn, &cs);
          const float x1 = acc[0][nc][r], x2 = acc[1][nc][r];
          acc[0][nc][r] = x1 * cs - x2 * sn;
          acc[1][nc][r] = x1 * sn + x2 * cs;
        }
      }
    }

    __syncthreads();   // LDS free (A-build / previous copy-out complete)
    if (mat < 2) {
      // Stage [token][channel], row-major Q/K.
      for (int hh = 0; hh < 2; ++hh)
        for (int nc = 0; nc < 4; ++nc)
#pragma unroll
          for (int r = 0; r < 8; ++r) {
            const int trow = r + 8 * group;
            const int ch   = (nc * 16 + mrow) * 16 + (h0 + hh);
            ldsO[trow * D_MODEL + ch] = (h16)acc[hh][nc][r];
          }
      __syncthreads();
      h16* o = (mat == 0 ? qo : ko) + ((size_t)(b * SEQ + t0)) * D_MODEL;
      for (int i = threadIdx.x; i < 16 * D_MODEL / 8; i += 256)
        *(v8h*)(o + (size_t)i * 8) = *(const v8h*)(ldsO + i * 8);
    } else {
      // Stage [channel][token] for transposed V.
      for (int hh = 0; hh < 2; ++hh)
        for (int nc = 0; nc < 4; ++nc)
#pragma unroll
          for (int r = 0; r < 8; ++r) {
            const int trow = r + 8 * group;
            const int ch   = (nc * 16 + mrow) * 16 + (h0 + hh);
            ldsO[ch * 16 + trow] = (h16)acc[hh][nc][r];
          }
      __syncthreads();
      for (int i = threadIdx.x; i < 2 * D_MODEL; i += 256) {
        const int ch = i >> 1, half = i & 1;
        *(v8h*)(vT + ((size_t)b * D_MODEL + ch) * SEQ + t0 + half * 8) =
            *(const v8h*)(ldsO + ch * 16 + half * 8);
      }
    }
  }
}

// ---------------------------------------------------------------------------
// Kernel 2: flash attention, D=1024, causal. Workgroup = 16 queries, 8 waves;
// wave w owns D-slice [128w, 128w+128). Per 32-key tile the TDM stages K
// (row-major) and a strided transposed-V tile into LDS; score partials are
// reduced cross-wave via ds_add_f32; both WMMA B operands are lane-contiguous.
// ---------------------------------------------------------------------------
__global__ __launch_bounds__(256) void attn_kernel(
    const h16* __restrict__ qf, const h16* __restrict__ kf,
    const h16* __restrict__ vT, float* __restrict__ y) {
  __shared__ h16   ldsK [32 * D_MODEL];   // 64 KB  [key][d]
  __shared__ h16   ldsVT[D_MODEL * 32];   // 64 KB  [d][key]
  __shared__ float ldsS [16 * 32];        // 2 KB score tile

  const int blk   = blockIdx.x;
  const int b     = blk / (SEQ / 16);
  const int q0    = (blk % (SEQ / 16)) * 16;
  const int lane  = threadIdx.x & 31;
  const int wave  = threadIdx.x >> 5;
  const int group = lane >> 4;
  const int mrow  = lane & 15;
  const int dbase = wave * 128;
  const float gamma = 0.03125f;          // 1/sqrt(1024)

  // Resident Q A-tiles: 4 x (16x32) f16 covering this wave's 128 dims.
  v16h aq[4];
  {
    const h16* qrow = qf + ((size_t)(b * SEQ + q0 + mrow)) * D_MODEL + dbase;
    for (int dc = 0; dc < 4; ++dc) {
      const v8h lo = *(const v8h*)(qrow + dc * 32 + group * 16);
      const v8h hi = *(const v8h*)(qrow + dc * 32 + group * 16 + 8);
      v16h a;
#pragma unroll
      for (int j = 0; j < 8; ++j) { a[j] = lo[j]; a[8 + j] = hi[j]; }
      aq[dc] = a;
    }
  }

  v8f Y[8];
  for (int nc = 0; nc < 8; ++nc)
    Y[nc] = (v8f){0.f, 0.f, 0.f, 0.f, 0.f, 0.f, 0.f, 0.f};
  float mstate = -3.0e38f, lstate = 0.0f;

  const int nT = (q0 + 16 + 31) / 32;
  for (int tile = 0; tile < nT; ++tile) {
    const int kv0 = tile * 32;
    if (wave == 0) {
      tdm_load_to_lds((unsigned)(uintptr_t)&ldsK[0],
                      kf + ((size_t)(b * SEQ + kv0)) * D_MODEL,
                      /*rows=*/32, /*elems=*/D_MODEL, /*stride=*/D_MODEL, 1);
      tdm_load_to_lds((unsigned)(uintptr_t)&ldsVT[0],
                      vT + ((size_t)b * D_MODEL) * SEQ + kv0,
                      /*rows=*/D_MODEL, /*elems=*/32, /*stride=*/SEQ, 1);
      __builtin_amdgcn_s_wait_tensorcnt(0);
    }
    ldsS[threadIdx.x]       = 0.0f;
    ldsS[256 + threadIdx.x] = 0.0f;
    __syncthreads();   // K/V staged AND score tile zeroed

    // Partial scores over this wave's 128 dims -> ds_add_f32 reduction.
    for (int kh = 0; kh < 2; ++kh) {
      v8f s = (v8f){0.f, 0.f, 0.f, 0.f, 0.f, 0.f, 0.f, 0.f};
      for (int dc = 0; dc < 4; ++dc) {
        const h16* krow = &ldsK[(size_t)(kh * 16 + mrow) * D_MODEL +
                                dbase + dc * 32 + group * 16];
        s = wmma_f16(aq[dc], *(const v16h*)krow, s);
      }
#pragma unroll
      for (int r = 0; r < 8; ++r)
        atomicAdd(&ldsS[(r + 8 * group) * 32 + kh * 16 + mrow], s[r]);
    }
    __syncthreads();   // full 16x32 scores ready

    // Online softmax (computed redundantly per wave; state replicated
    // across lane pairs (L, L+16), row = lane%16).
    const int qidx = q0 + mrow;
    float sj[16];
    float tmax = -3.0e38f;
#pragma unroll
    for (int j = 0; j < 16; ++j) {
      const int K = group * 16 + j;
      const float sv = ldsS[mrow * 32 + K] * gamma;
      sj[j] = ((kv0 + K) > qidx) ? -3.0e38f : sv;
      tmax  = fmaxf(tmax, sj[j]);
    }
    tmax = fmaxf(tmax, __shfl_xor(tmax, 16, 32));
    const float mnew    = fmaxf(mstate, tmax);
    const float rescale = __expf(mstate - mnew);
    float tsum = 0.0f;
    v16h pa;
#pragma unroll
    for (int j = 0; j < 16; ++j) {
      const float p = (sj[j] < -1.0e38f) ? 0.0f : __expf(sj[j] - mnew);
      tsum += p;
      pa[j] = (h16)p;
    }
    tsum  += __shfl_xor(tsum, 16, 32);
    lstate = lstate * rescale + tsum;
    mstate = mnew;

    // Rescale running output (row of Y element = r + 8*group).
#pragma unroll
    for (int r = 0; r < 8; ++r) {
      const float sc = __shfl(rescale, r + 8 * group, 32);
#pragma unroll
      for (int nc = 0; nc < 8; ++nc) Y[nc][r] *= sc;
    }

    // Y += P(16x32) * V(32x16 per 16-col chunk); B from transposed-V LDS,
    // lane-contiguous: slot j -> key = group*16 + j at fixed d.
    for (int nc = 0; nc < 8; ++nc) {
      const h16* vrow = &ldsVT[(size_t)(dbase + nc * 16 + mrow) * 32 + group * 16];
      Y[nc] = wmma_f16(pa, *(const v16h*)vrow, Y[nc]);
    }
    __syncthreads();   // before next tile overwrites LDS
  }

  // Normalize and store f32 output.
#pragma unroll
  for (int r = 0; r < 8; ++r) {
    const int   M    = r + 8 * group;
    const float linv = 1.0f / __shfl(lstate, M, 32);
    float* yrow = y + ((size_t)(b * SEQ + q0 + M)) * D_MODEL + dbase;
    for (int nc = 0; nc < 8; ++nc)
      yrow[nc * 16 + mrow] = Y[nc][r] * linv;
  }
}

extern "C" void kernel_launch(void* const* d_in, const int* in_sizes, int n_in,
                              void* d_out, int out_size, void* d_ws, size_t ws_size,
                              hipStream_t stream) {
  const float* x  = (const float*)d_in[0];
  const float* wq = (const float*)d_in[1];
  const float* wk = (const float*)d_in[2];
  const float* wv = (const float*)d_in[3];
  float* yo = (float*)d_out;

  h16* qf = (h16*)d_ws;                                  // 32 MB  [b][t][d]
  h16* kf = qf + (size_t)BATCH * SEQ * D_MODEL;          // 32 MB  [b][t][d]
  h16* vT = kf + (size_t)BATCH * SEQ * D_MODEL;          // 32 MB  [b][d][t]
  h16* wp = vT + (size_t)BATCH * SEQ * D_MODEL;          // 384 KB packed weights

  pack_w_kernel<<<dim3(768), 256, 0, stream>>>(wq, wk, wv, wp);
  dim3 grid(BATCH * SEQ / 16);
  qkv_rope_kernel<<<grid, 256, 0, stream>>>(x, wp, qf, kf, vT);
  attn_kernel<<<grid, 256, 0, stream>>>(qf, kf, vT, yo);
}
